// Net_10685878633098
// MI455X (gfx1250) — matrix operand order repaired and checked by reading, not compiled, not run on hardware
//
#include <hip/hip_runtime.h>
#include <hip/hip_bf16.h>

#define N_NODES 50000
#define N_EDGES 800000
#define BN_EPS 1e-5f
#define MTILES (N_NODES / 16)   // 3125, exact

typedef __attribute__((ext_vector_type(16))) _Float16 v16h;
typedef __attribute__((ext_vector_type(8)))  _Float16 v8h;
typedef __attribute__((ext_vector_type(8)))  float    v8f;
typedef __attribute__((ext_vector_type(4)))  float    v4f;

// ---------------- degree / normalization ----------------

__global__ void k_init_deg(float* __restrict__ deg) {
    int i = blockIdx.x * blockDim.x + threadIdx.x;
    if (i < N_NODES) deg[i] = 1.0f;   // self-loop contribution
}

__global__ void k_deg(const int* __restrict__ dst, float* __restrict__ deg) {
    int e = blockIdx.x * blockDim.x + threadIdx.x;
    if (e < N_EDGES) atomicAdd(&deg[dst[e]], 1.0f);
}

__global__ void k_dinv_self(const float* __restrict__ x, const float* __restrict__ deg,
                            float* __restrict__ dinv, float* __restrict__ agg1) {
    int i = blockIdx.x * blockDim.x + threadIdx.x;
    if (i < N_NODES) {
        float dv = rsqrtf(deg[i]);          // deg >= 1 always (self loops)
        dinv[i]  = dv;
        agg1[i]  = x[i] * dv * dv;          // self-loop message (scalar feature)
    }
}

// conv1 edge aggregation on the SCALAR pre-expansion feature; also cache norm[e]
__global__ void k_agg1(const int* __restrict__ src, const int* __restrict__ dst,
                       const float* __restrict__ x, const float* __restrict__ dinv,
                       float* __restrict__ norm, float* __restrict__ agg1) {
    int e = blockIdx.x * blockDim.x + threadIdx.x;
    if (e < N_EDGES) {
        int s = src[e], d = dst[e];
        float nv = dinv[s] * dinv[d];
        norm[e] = nv;
        atomicAdd(&agg1[d], x[s] * nv);
    }
}

// h1[i,j] = f16( relu( BN1( agg1[i]*W1[j] + b1[j] ) ) ),  [N,256] row-major
__global__ void k_h1(const float* __restrict__ agg1, const float* __restrict__ W1,
                     const float* __restrict__ b1, const float* __restrict__ g1,
                     const float* __restrict__ be1, const float* __restrict__ rm1,
                     const float* __restrict__ rv1, _Float16* __restrict__ h1) {
    int i = blockIdx.x;          // node
    int j = threadIdx.x;         // channel 0..255
    float v = agg1[i] * W1[j] + b1[j];
    v = (v - rm1[j]) * rsqrtf(rv1[j] + BN_EPS) * g1[j] + be1[j];
    v = fmaxf(v, 0.0f);
    h1[(size_t)i * 256 + j] = (_Float16)v;
}

// Pre-pack W2 [256,64] f32 into per-lane WMMA B-fragment layout (f16):
//   w2p[ ((kk*4 + nt)*32 + lane)*16 + q ]  =  W2[K, N]
//   K = kk*32 + (lane>>4)*16 + q,  N = nt*16 + (lane&15)
// so each lane's 16 halves of a fragment are contiguous (two b128 loads).
__global__ void k_w2p(const float* __restrict__ W2, _Float16* __restrict__ w2p) {
    int t = blockIdx.x * blockDim.x + threadIdx.x;   // 0 .. 16383
    if (t >= 8 * 4 * 32 * 16) return;
    int q  = t & 15;
    int L  = (t >> 4) & 31;
    int nt = (t >> 9) & 3;
    int kk = t >> 11;
    int K = kk * 32 + (L >> 4) * 16 + q;
    int N = nt * 16 + (L & 15);
    w2p[t] = (_Float16)W2[K * 64 + N];
}

// ---------------- conv2 GEMM via WMMA:  z = h1 @ W2  ----------------
// one wave per 16-row M-tile; epilogue writes z and agg2 = z * dinv^2 (self loop)
__global__ __launch_bounds__(256) void k_gemm(
    const _Float16* __restrict__ h1, const _Float16* __restrict__ w2p,
    const float* __restrict__ dinv,
    float* __restrict__ z, float* __restrict__ agg2)
{
    const int wid   = threadIdx.x >> 5;
    const int lane  = threadIdx.x & 31;
    const int mtile = blockIdx.x * 8 + wid;
    if (mtile >= MTILES) return;          // wave-uniform exit: EXEC stays all-1 inside

    const int lrow = lane & 15;
    const int hi   = lane >> 4;

    v8f acc[4] = {};                       // 4 N-tiles of 16 cols (N = 64)

    const _Float16* __restrict__ abase = h1 + (size_t)(mtile * 16 + lrow) * 256;
    const _Float16* __restrict__ bbase = w2p + (size_t)lane * 16;
#pragma unroll
    for (int kk = 0; kk < 8; ++kk) {       // K = 256 in steps of 32
        const int k0 = kk * 32;
        // A fragment 16x32 f16 (contiguous, matches w2p): lanes 0-15 K{0-15}, lanes 16-31 K{16-31}
        union { v16h v; v8h h[2]; } a;
        a.h[0] = *(const v8h*)(abase + k0 + hi * 16);
        a.h[1] = *(const v8h*)(abase + k0 + hi * 16 + 8);
#pragma unroll
        for (int nt = 0; nt < 4; ++nt) {
            // B fragment: pre-packed, contiguous per lane -> two b128 loads
            union { v16h v; v8h h[2]; } b;
            const _Float16* bf = bbase + (size_t)(kk * 4 + nt) * 32 * 16;
            b.h[0] = *(const v8h*)(bf);
            b.h[1] = *(const v8h*)(bf + 8);
            acc[nt] = __builtin_amdgcn_wmma_f32_16x16x32_f16(
                false, a.v, false, b.v, (short)0, acc[nt], false, false);
        }
    }

    // epilogue: C layout -> row = hi*8 + r, col = nt*16 + lrow
    float d2[8];
#pragma unroll
    for (int r = 0; r < 8; ++r) {
        float dv = dinv[mtile * 16 + hi * 8 + r];
        d2[r] = dv * dv;
    }
#pragma unroll
    for (int nt = 0; nt < 4; ++nt) {
        const int c = nt * 16 + lrow;
#pragma unroll
        for (int r = 0; r < 8; ++r) {
            const int node = mtile * 16 + hi * 8 + r;
            float v = acc[nt][r];
            z[(size_t)node * 64 + c]    = v;
            agg2[(size_t)node * 64 + c] = v * d2[r];   // self-loop init
        }
    }
}

// conv2 edge aggregation, 64-wide: agg2[d,:] += norm[e] * z[s,:]
// 4 channels per thread: one float4 gather + 4 atomics, 4x less edge metadata traffic
__global__ void k_agg2(const int* __restrict__ src, const int* __restrict__ dst,
                       const float* __restrict__ norm, const float* __restrict__ z,
                       float* __restrict__ agg2) {
    long long idx = (long long)blockIdx.x * blockDim.x + threadIdx.x;
    if (idx >= (long long)N_EDGES * 16) return;
    int e  = (int)(idx >> 4);
    int c4 = (int)(idx & 15) * 4;
    int s = src[e], d = dst[e];
    float nv = norm[e];
    v4f zv = *(const v4f*)(z + (size_t)s * 64 + c4);
    float* ap = agg2 + (size_t)d * 64 + c4;
    atomicAdd(ap + 0, nv * zv.x);
    atomicAdd(ap + 1, nv * zv.y);
    atomicAdd(ap + 2, nv * zv.z);
    atomicAdd(ap + 3, nv * zv.w);
}

// ---------------- head: BN2+ReLU, 64->16 ReLU, 16->6, log_softmax ----------------
__global__ void k_head(const float* __restrict__ agg2,
                       const float* __restrict__ b2, const float* __restrict__ g2,
                       const float* __restrict__ be2, const float* __restrict__ rm2,
                       const float* __restrict__ rv2,
                       const float* __restrict__ lw1, const float* __restrict__ lb1,
                       const float* __restrict__ lw2, const float* __restrict__ lb2,
                       float* __restrict__ out) {
    int i = blockIdx.x * blockDim.x + threadIdx.x;
    if (i >= N_NODES) return;
    const float* __restrict__ zrow = agg2 + (size_t)i * 64;

    float u[16];
#pragma unroll
    for (int j = 0; j < 16; ++j) u[j] = lb1[j];
#pragma unroll
    for (int k = 0; k < 64; ++k) {
        float hv = (zrow[k] + b2[k] - rm2[k]) * rsqrtf(rv2[k] + BN_EPS) * g2[k] + be2[k];
        hv = fmaxf(hv, 0.0f);
#pragma unroll
        for (int j = 0; j < 16; ++j) u[j] = fmaf(hv, lw1[k * 16 + j], u[j]);
    }
#pragma unroll
    for (int j = 0; j < 16; ++j) u[j] = fmaxf(u[j], 0.0f);

    float t6[6];
#pragma unroll
    for (int c = 0; c < 6; ++c) t6[c] = lb2[c];
#pragma unroll
    for (int j = 0; j < 16; ++j) {
#pragma unroll
        for (int c = 0; c < 6; ++c) t6[c] = fmaf(u[j], lw2[j * 6 + c], t6[c]);
    }
    float m = t6[0];
#pragma unroll
    for (int c = 1; c < 6; ++c) m = fmaxf(m, t6[c]);
    float sum = 0.0f;
#pragma unroll
    for (int c = 0; c < 6; ++c) sum += expf(t6[c] - m);
    float lse = logf(sum);
#pragma unroll
    for (int c = 0; c < 6; ++c) out[(size_t)i * 6 + c] = t6[c] - m - lse;
}

// ---------------- launcher ----------------

extern "C" void kernel_launch(void* const* d_in, const int* in_sizes, int n_in,
                              void* d_out, int out_size, void* d_ws, size_t ws_size,
                              hipStream_t stream) {
    const float* x   = (const float*)d_in[0];
    const float* W1  = (const float*)d_in[1];
    const float* b1  = (const float*)d_in[2];
    const float* g1  = (const float*)d_in[3];
    const float* be1 = (const float*)d_in[4];
    const float* rm1 = (const float*)d_in[5];
    const float* rv1 = (const float*)d_in[6];
    const float* W2  = (const float*)d_in[7];
    const float* b2  = (const float*)d_in[8];
    const float* g2  = (const float*)d_in[9];
    const float* be2 = (const float*)d_in[10];
    const float* rm2 = (const float*)d_in[11];
    const float* rv2 = (const float*)d_in[12];
    const float* lw1 = (const float*)d_in[13];
    const float* lb1 = (const float*)d_in[14];
    const float* lw2 = (const float*)d_in[15];
    const float* lb2 = (const float*)d_in[16];
    const int* edge  = (const int*)d_in[17];
    const int* src = edge;             // edge_index[0]
    const int* dst = edge + N_EDGES;   // edge_index[1]
    float* out = (float*)d_out;

    // workspace carve (harness poisons to 0xAA; we init everything we read)
    char* ws = (char*)d_ws;
    size_t off = 0;
    auto take = [&](size_t bytes) -> void* {
        void* p = ws + off;
        off += (bytes + 255) & ~(size_t)255;
        return p;
    };
    float*    deg  = (float*)take((size_t)N_NODES * 4);
    float*    dinv = (float*)take((size_t)N_NODES * 4);
    float*    agg1 = (float*)take((size_t)N_NODES * 4);
    float*    norm = (float*)take((size_t)N_EDGES * 4);
    _Float16* w2p  = (_Float16*)take((size_t)8 * 4 * 32 * 16 * 2);
    _Float16* h1   = (_Float16*)take((size_t)N_NODES * 256 * 2);
    float*    z    = (float*)take((size_t)N_NODES * 64 * 4);
    float*    agg2 = (float*)take((size_t)N_NODES * 64 * 4);
    (void)ws_size; (void)n_in; (void)in_sizes; (void)out_size;

    k_init_deg<<<(N_NODES + 255) / 256, 256, 0, stream>>>(deg);
    k_deg<<<(N_EDGES + 255) / 256, 256, 0, stream>>>(dst, deg);
    k_dinv_self<<<(N_NODES + 255) / 256, 256, 0, stream>>>(x, deg, dinv, agg1);
    k_agg1<<<(N_EDGES + 255) / 256, 256, 0, stream>>>(src, dst, x, dinv, norm, agg1);
    k_w2p<<<(8 * 4 * 32 * 16 + 255) / 256, 256, 0, stream>>>(W2, w2p);
    k_h1<<<N_NODES, 256, 0, stream>>>(agg1, W1, b1, g1, be1, rm1, rv1, h1);
    k_gemm<<<(MTILES + 7) / 8, 256, 0, stream>>>(h1, w2p, dinv, z, agg2);
    {
        long long total = (long long)N_EDGES * 16;
        int blocks = (int)((total + 255) / 256);
        k_agg2<<<blocks, 256, 0, stream>>>(src, dst, norm, z, agg2);
    }
    k_head<<<(N_NODES + 255) / 256, 256, 0, stream>>>(
        agg2, b2, g2, be2, rm2, rv2, lw1, lb1, lw2, lb2, out);
}
